// QuantizableWormholeAttention_14972255994629
// MI455X (gfx1250) — matrix operand order repaired and checked by reading, not compiled, not run on hardware
//
#include <hip/hip_runtime.h>
#include <hip/hip_bf16.h>

#define B_  4
#define S_  2048
#define H_  768
#define NH_ 12
#define DH_ 64
#define WH_ 128

typedef __attribute__((ext_vector_type(16))) __bf16        v16bf;
typedef __attribute__((ext_vector_type(8)))  float         v8f;
typedef __attribute__((ext_vector_type(8)))  unsigned short u16x8;
typedef __attribute__((ext_vector_type(16))) unsigned short u16x16;
typedef __attribute__((ext_vector_type(4)))  unsigned short u16x4;

enum { EP_F32 = 0, EP_HEADS = 1, EP_HEADS_T = 2 };

static __device__ __forceinline__ unsigned short f2bf(float f) {
  unsigned u = __builtin_bit_cast(unsigned, f);
  u += 0x7fffu + ((u >> 16) & 1u);            // round-to-nearest-even
  return (unsigned short)(u >> 16);
}

// A fragment: 16x32 bf16, row-major source, K-contiguous.
// Lane L (L<16): row=L, elems 0..7 = K[k0+0..7],  8..15 = K[k0+16..23]
// Lane L (>=16): row=L-16, elems 0..7 = K[k0+8..15], 8..15 = K[k0+24..31]
static __device__ __forceinline__ v16bf load_a_frag(const unsigned short* A, int lda,
                                                    int row0, int k0) {
  int lane = threadIdx.x & 31;
  int r = lane & 15, half = lane >> 4;
  const unsigned short* p = A + (size_t)(row0 + r) * lda + k0 + half * 8;
  u16x8 lo = *(const u16x8*)p;
  u16x8 hi = *(const u16x8*)(p + 16);
  u16x16 t = __builtin_shufflevector(lo, hi, 0,1,2,3,4,5,6,7,8,9,10,11,12,13,14,15);
  return __builtin_bit_cast(v16bf, t);
}

// B fragment: 32x16 bf16 from BT stored row-major [N][K] (i.e. B column-major).
// Lane L (L<16): col=L, elems 0..7 = K[k0+0..7], 8..15 = K[k0+16..23]
// Lane L (>=16): col=L-16, elems 0..7 = K[k0+8..15], 8..15 = K[k0+24..31]
static __device__ __forceinline__ v16bf load_bt_frag(const unsigned short* BT, int ldb,
                                                     int col0, int k0) {
  int lane = threadIdx.x & 31;
  int n = lane & 15, half = lane >> 4;
  const unsigned short* p = BT + (size_t)(col0 + n) * ldb + k0 + half * 8;
  u16x8 lo = *(const u16x8*)p;
  u16x8 hi = *(const u16x8*)(p + 16);
  u16x16 t = __builtin_shufflevector(lo, hi, 0,1,2,3,4,5,6,7,8,9,10,11,12,13,14,15);
  return __builtin_bit_cast(v16bf, t);
}

static __device__ __forceinline__ v8f wmma_bf16(v16bf a, v16bf b, v8f c) {
  return __builtin_amdgcn_wmma_f32_16x16x32_bf16(false, a, false, b, (short)0, c,
                                                 false, false);
}

// ---------------- elementwise conversion ----------------
__global__ void __launch_bounds__(256)
f32_to_bf16_kernel(const float* __restrict__ in, unsigned short* __restrict__ out, size_t n) {
  size_t i = (size_t)blockIdx.x * blockDim.x + threadIdx.x;
  if (i < n) out[i] = f2bf(in[i]);
}

// WT[n*K + k] = bf16(W[k*N + n])
__global__ void __launch_bounds__(256)
transpose_bf16_kernel(const float* __restrict__ W, unsigned short* __restrict__ WT,
                      int K, int N) {
  size_t i = (size_t)blockIdx.x * blockDim.x + threadIdx.x;
  if (i < (size_t)K * N) {
    int n = (int)(i / K), k = (int)(i % K);
    WT[i] = f2bf(W[(size_t)k * N + n]);
  }
}

// ---------------- wormhole gate MLP ----------------
// one 128-thread block per token row: H->WH, LN, ReLU, WH->NH, sigmoid
__global__ void __launch_bounds__(WH_)
gate_kernel(const float* __restrict__ hidden, const float* __restrict__ Wg1,
            const float* __restrict__ bg1, const float* __restrict__ gw,
            const float* __restrict__ gb, const float* __restrict__ Wg2,
            const float* __restrict__ bg2, float* __restrict__ gates) {
  int r = blockIdx.x, t = threadIdx.x;
  __shared__ float hrow[H_];
  __shared__ float red[WH_];
  __shared__ float gsh[WH_];
  const float* hp = hidden + (size_t)r * H_;
  for (int i = t; i < H_; i += WH_) hrow[i] = hp[i];
  __syncthreads();
  float acc = bg1[t];
  for (int k = 0; k < H_; ++k) acc += hrow[k] * Wg1[(size_t)k * WH_ + t];
  red[t] = acc; __syncthreads();
  for (int s = WH_ / 2; s > 0; s >>= 1) { if (t < s) red[t] += red[t + s]; __syncthreads(); }
  float mu = red[0] / (float)WH_; __syncthreads();
  float d = acc - mu;
  red[t] = d * d; __syncthreads();
  for (int s = WH_ / 2; s > 0; s >>= 1) { if (t < s) red[t] += red[t + s]; __syncthreads(); }
  float var = red[0] / (float)WH_; __syncthreads();
  float gn = d * rsqrtf(var + 1e-12f) * gw[t] + gb[t];
  gsh[t] = fmaxf(gn, 0.f);
  __syncthreads();
  if (t < NH_) {
    float s = bg2[t];
    for (int j = 0; j < WH_; ++j) s += gsh[j] * Wg2[j * NH_ + t];
    gates[(size_t)r * NH_ + t] = 1.f / (1.f + __expf(-s));
  }
}

// ---------------- generic bf16 WMMA GEMM ----------------
// 8 waves/block, each wave one 16x16 tile; block covers 128 rows x 16 cols.
__global__ void __launch_bounds__(256)
gemm_bf16_kernel(const unsigned short* __restrict__ A, const unsigned short* __restrict__ BT,
                 const float* __restrict__ bias, void* __restrict__ out,
                 int N, int Kdim, int lda, int ldb, int mode) {
  int wave = threadIdx.x >> 5;
  int row0 = (blockIdx.x * 8 + wave) * 16;
  int col0 = blockIdx.y * 16;
  v8f acc = {0.f, 0.f, 0.f, 0.f, 0.f, 0.f, 0.f, 0.f};
  for (int k0 = 0; k0 < Kdim; k0 += 32) {
    v16bf a = load_a_frag(A, lda, row0, k0);
    v16bf b = load_bt_frag(BT, ldb, col0, k0);
    acc = wmma_bf16(a, b, acc);
  }
  int lane = threadIdx.x & 31;
  int n = col0 + (lane & 15);
  int mbase = row0 + 8 * (lane >> 4);
  float bv = bias ? bias[n] : 0.f;
#pragma unroll
  for (int i = 0; i < 8; ++i) {
    int m = mbase + i;
    float v = acc[i] + bv;
    if (mode == EP_F32) {
      ((float*)out)[(size_t)m * N + n] = v;
    } else {
      int b_ = m / S_, s = m % S_, h = n / DH_, d = n % DH_;
      if (mode == EP_HEADS)   // [b][h][s][d]
        ((unsigned short*)out)[(((size_t)(b_ * NH_ + h)) * S_ + s) * DH_ + d] = f2bf(v);
      else                    // EP_HEADS_T: [b][h][d][s]
        ((unsigned short*)out)[(((size_t)(b_ * NH_ + h)) * DH_ + d) * S_ + s] = f2bf(v);
    }
  }
}

// ---------------- attention scores: (Q K^T)/8 * gate + mask ----------------
__global__ void __launch_bounds__(256)
scores_kernel(const unsigned short* __restrict__ Q, const unsigned short* __restrict__ K,
              const float* __restrict__ gates, const float* __restrict__ amask,
              float* __restrict__ probs) {
  int bh = blockIdx.z;
  int b = bh / NH_, h = bh % NH_;
  int wave = threadIdx.x >> 5;
  int q0 = blockIdx.x * 16;
  int k0 = (blockIdx.y * 8 + wave) * 16;
  const unsigned short* Qb = Q + (size_t)bh * S_ * DH_;
  const unsigned short* Kb = K + (size_t)bh * S_ * DH_;
  v8f acc = {0.f, 0.f, 0.f, 0.f, 0.f, 0.f, 0.f, 0.f};
#pragma unroll
  for (int d0 = 0; d0 < DH_; d0 += 32)
    acc = wmma_bf16(load_a_frag(Qb, DH_, q0, d0), load_bt_frag(Kb, DH_, k0, d0), acc);
  int lane = threadIdx.x & 31;
  int n = k0 + (lane & 15);
  int mbase = q0 + 8 * (lane >> 4);
  float madd = (1.f - amask[(size_t)b * S_ + n]) * -10000.f;
  float* prow = probs + (size_t)bh * S_ * S_;
#pragma unroll
  for (int i = 0; i < 8; ++i) {
    int m = mbase + i;
    float g = gates[((size_t)b * S_ + m) * NH_ + h];
    prow[(size_t)m * S_ + n] = acc[i] * 0.125f * g + madd;
  }
}

// ---------------- row softmax (S=2048), also emit bf16 copy ----------------
__global__ void __launch_bounds__(256)
softmax_kernel(float* __restrict__ probs, unsigned short* __restrict__ Pbf) {
  size_t row = blockIdx.x;
  int t = threadIdx.x;
  float4* rp = (float4*)(probs + row * S_);
  float4 x0 = rp[t], x1 = rp[t + 256];
  __shared__ float red[256];
  float mx = fmaxf(fmaxf(fmaxf(x0.x, x0.y), fmaxf(x0.z, x0.w)),
                   fmaxf(fmaxf(x1.x, x1.y), fmaxf(x1.z, x1.w)));
  red[t] = mx; __syncthreads();
  for (int s = 128; s > 0; s >>= 1) { if (t < s) red[t] = fmaxf(red[t], red[t + s]); __syncthreads(); }
  float rmax = red[0]; __syncthreads();
  x0.x = __expf(x0.x - rmax); x0.y = __expf(x0.y - rmax);
  x0.z = __expf(x0.z - rmax); x0.w = __expf(x0.w - rmax);
  x1.x = __expf(x1.x - rmax); x1.y = __expf(x1.y - rmax);
  x1.z = __expf(x1.z - rmax); x1.w = __expf(x1.w - rmax);
  float sum = x0.x + x0.y + x0.z + x0.w + x1.x + x1.y + x1.z + x1.w;
  red[t] = sum; __syncthreads();
  for (int s = 128; s > 0; s >>= 1) { if (t < s) red[t] += red[t + s]; __syncthreads(); }
  float inv = 1.f / red[0];
  x0.x *= inv; x0.y *= inv; x0.z *= inv; x0.w *= inv;
  x1.x *= inv; x1.y *= inv; x1.z *= inv; x1.w *= inv;
  rp[t] = x0; rp[t + 256] = x1;
  u16x4 y0 = {f2bf(x0.x), f2bf(x0.y), f2bf(x0.z), f2bf(x0.w)};
  u16x4 y1 = {f2bf(x1.x), f2bf(x1.y), f2bf(x1.z), f2bf(x1.w)};
  u16x4* bp = (u16x4*)(Pbf + row * S_);
  bp[t] = y0; bp[t + 256] = y1;
}

// ---------------- ctx = P @ V (V stored transposed [b][h][d][s]) ----------------
__global__ void __launch_bounds__(256)
pv_kernel(const unsigned short* __restrict__ Pbf, const unsigned short* __restrict__ Vt,
          unsigned short* __restrict__ ctx) {
  int bh = blockIdx.z;
  int b = bh / NH_, h = bh % NH_;
  int wave = threadIdx.x >> 5;
  int row0 = (blockIdx.x * 8 + wave) * 16;
  int col0 = blockIdx.y * 16;
  const unsigned short* Ab = Pbf + (size_t)bh * S_ * S_;
  const unsigned short* Bt = Vt + (size_t)bh * DH_ * S_;
  v8f acc = {0.f, 0.f, 0.f, 0.f, 0.f, 0.f, 0.f, 0.f};
  for (int k0 = 0; k0 < S_; k0 += 32)
    acc = wmma_bf16(load_a_frag(Ab, S_, row0, k0), load_bt_frag(Bt, S_, col0, k0), acc);
  int lane = threadIdx.x & 31;
  int d = col0 + (lane & 15);
  int mbase = row0 + 8 * (lane >> 4);
#pragma unroll
  for (int i = 0; i < 8; ++i) {
    int m = mbase + i;
    ctx[((size_t)b * S_ + m) * H_ + h * DH_ + d] = f2bf(acc[i]);
  }
}

// ---------------- residual + layernorm ----------------
__global__ void __launch_bounds__(256)
ln_residual_kernel(const float* __restrict__ proj, const float* __restrict__ hidden,
                   const float* __restrict__ w, const float* __restrict__ bp,
                   float* __restrict__ out) {
  int r = blockIdx.x, t = threadIdx.x;
  const float* pp = proj + (size_t)r * H_;
  const float* hp = hidden + (size_t)r * H_;
  __shared__ float red[256];
  float x[3]; float s = 0.f;
#pragma unroll
  for (int j = 0; j < 3; ++j) { int idx = t + j * 256; x[j] = pp[idx] + hp[idx]; s += x[j]; }
  red[t] = s; __syncthreads();
  for (int k = 128; k > 0; k >>= 1) { if (t < k) red[t] += red[t + k]; __syncthreads(); }
  float mu = red[0] / (float)H_; __syncthreads();
  s = 0.f;
#pragma unroll
  for (int j = 0; j < 3; ++j) { float d = x[j] - mu; s += d * d; }
  red[t] = s; __syncthreads();
  for (int k = 128; k > 0; k >>= 1) { if (t < k) red[t] += red[t + k]; __syncthreads(); }
  float rstd = rsqrtf(red[0] / (float)H_ + 1e-12f);
#pragma unroll
  for (int j = 0; j < 3; ++j) {
    int idx = t + j * 256;
    out[(size_t)r * H_ + idx] = (x[j] - mu) * rstd * w[idx] + bp[idx];
  }
}

extern "C" void kernel_launch(void* const* d_in, const int* in_sizes, int n_in,
                              void* d_out, int out_size, void* d_ws, size_t ws_size,
                              hipStream_t stream) {
  (void)in_sizes; (void)n_in; (void)out_size; (void)ws_size;
  const float* hidden = (const float*)d_in[0];
  const float* amask  = (const float*)d_in[1];
  const float* Wq  = (const float*)d_in[2];  const float* bq  = (const float*)d_in[3];
  const float* Wk  = (const float*)d_in[4];  const float* bk  = (const float*)d_in[5];
  const float* Wv  = (const float*)d_in[6];  const float* bv  = (const float*)d_in[7];
  const float* Wg1 = (const float*)d_in[8];  const float* bg1 = (const float*)d_in[9];
  const float* glnw = (const float*)d_in[10]; const float* glnb = (const float*)d_in[11];
  const float* Wg2 = (const float*)d_in[12]; const float* bg2 = (const float*)d_in[13];
  const float* Wo  = (const float*)d_in[14]; const float* bo  = (const float*)d_in[15];
  const float* lnw = (const float*)d_in[16]; const float* lnb = (const float*)d_in[17];

  float* outp  = (float*)d_out;                          // [B,S,H]
  float* probs = outp + (size_t)B_ * S_ * H_;            // [B,NH,S,S]

  char* w = (char*)d_ws;
  auto alloc = [&](size_t bytes) -> char* {
    char* p = w; w += (bytes + 255) & ~(size_t)255; return p;
  };
  unsigned short* hbf  = (unsigned short*)alloc((size_t)B_ * S_ * H_ * 2);
  unsigned short* WqT  = (unsigned short*)alloc((size_t)H_ * H_ * 2);
  unsigned short* WkT  = (unsigned short*)alloc((size_t)H_ * H_ * 2);
  unsigned short* WvT  = (unsigned short*)alloc((size_t)H_ * H_ * 2);
  unsigned short* WoT  = (unsigned short*)alloc((size_t)H_ * H_ * 2);
  unsigned short* Qb   = (unsigned short*)alloc((size_t)B_ * S_ * H_ * 2);
  unsigned short* Kb   = (unsigned short*)alloc((size_t)B_ * S_ * H_ * 2);
  unsigned short* Vtb  = (unsigned short*)alloc((size_t)B_ * S_ * H_ * 2);
  float*          gates = (float*)alloc((size_t)B_ * S_ * NH_ * 4);
  unsigned short* Pbf  = (unsigned short*)alloc((size_t)B_ * NH_ * S_ * S_ * 2);
  unsigned short* ctxb = (unsigned short*)alloc((size_t)B_ * S_ * H_ * 2);
  float*          proj = (float*)alloc((size_t)B_ * S_ * H_ * 4);

  size_t nh = (size_t)B_ * S_ * H_;
  f32_to_bf16_kernel<<<(unsigned)((nh + 255) / 256), 256, 0, stream>>>(hidden, hbf, nh);
  size_t nw = (size_t)H_ * H_;
  unsigned twb = (unsigned)((nw + 255) / 256);
  transpose_bf16_kernel<<<twb, 256, 0, stream>>>(Wq, WqT, H_, H_);
  transpose_bf16_kernel<<<twb, 256, 0, stream>>>(Wk, WkT, H_, H_);
  transpose_bf16_kernel<<<twb, 256, 0, stream>>>(Wv, WvT, H_, H_);
  transpose_bf16_kernel<<<twb, 256, 0, stream>>>(Wo, WoT, H_, H_);

  gate_kernel<<<B_ * S_, WH_, 0, stream>>>(hidden, Wg1, bg1, glnw, glnb, Wg2, bg2, gates);

  dim3 gg(B_ * S_ / 128, H_ / 16);
  gemm_bf16_kernel<<<gg, 256, 0, stream>>>(hbf, WqT, bq, Qb,  H_, H_, H_, H_, EP_HEADS);
  gemm_bf16_kernel<<<gg, 256, 0, stream>>>(hbf, WkT, bk, Kb,  H_, H_, H_, H_, EP_HEADS);
  gemm_bf16_kernel<<<gg, 256, 0, stream>>>(hbf, WvT, bv, Vtb, H_, H_, H_, H_, EP_HEADS_T);

  dim3 gs(S_ / 16, S_ / 128, B_ * NH_);
  scores_kernel<<<gs, 256, 0, stream>>>(Qb, Kb, gates, amask, probs);

  softmax_kernel<<<B_ * NH_ * S_, 256, 0, stream>>>(probs, Pbf);

  dim3 gp(S_ / 128, DH_ / 16, B_ * NH_);
  pv_kernel<<<gp, 256, 0, stream>>>(Pbf, Vtb, ctxb);

  gemm_bf16_kernel<<<gg, 256, 0, stream>>>(ctxb, WoT, bo, proj, H_, H_, H_, H_, EP_F32);

  ln_residual_kernel<<<B_ * S_, 256, 0, stream>>>(proj, hidden, lnw, lnb, outp);
}
